// BinarizedConv2d_5935644803402
// MI455X (gfx1250) — matrix operand order repaired and checked by reading, not compiled, not run on hardware
//
#include <hip/hip_runtime.h>
#include <hip/hip_bf16.h>

// BinarizedConv2d as implicit GEMM on CDNA5 WMMA (bf16 -> f32).
// D[oc][pix] = sum_k' sign(W)[oc][k'] * im2col(X)[k'][pix]
// K ordered as k' = (kh*3+kw)*256 + ic: each 32-wide K slice has a fixed conv
// tap (kh,kw), so the padding predicate + tap address are slice-invariant.
// M=256 (oc), N=100352 (pix = n*3136 + oh*56 + ow), K=2304.
//
// - Weights sign-binarized + K-reordered ONCE into d_ws as bf16 [oc][k']
//   (1.13 MB, L2-resident): main-loop weight staging = two b128 loads.
// - Software-pipelined main loop: ping-pong LDS buffers, ONE barrier per
//   slice; next slice's global loads issue before the WMMA block so VMEM
//   latency overlaps matrix ops.

#define IN_CH   256
#define OUT_CH  256
#define H       56
#define W       56
#define NB      32
#define KK      2304          // 256*9
#define PIX     (H * W)       // 3136
#define NPIX    (NB * PIX)    // 100352

#define TILE_OC  128          // M tile per workgroup
#define TILE_PX  128          // N tile per workgroup
#define KSTEP    32           // WMMA K for bf16
#define NSLICE   (KK / KSTEP) // 72 = 9 taps * 8 ic-slices

#define LDS_STRIDE 40         // 32 bf16 payload + 8 pad -> 80B rows (bank-conflict free)

typedef __attribute__((ext_vector_type(16))) __bf16          v16bf;
typedef __attribute__((ext_vector_type(8)))  float           v8f;
typedef __attribute__((ext_vector_type(8)))  unsigned short  v8u16;
typedef __attribute__((ext_vector_type(16))) unsigned short  v16u16;

union FragAB {
    v16u16 u;
    v16bf  b;
    v8u16  h[2];
};

__device__ __forceinline__ unsigned short f32_to_bf16(float f) {
    unsigned int u = __float_as_uint(f);
    u += 0x7FFFu + ((u >> 16) & 1u);          // round to nearest even
    return (unsigned short)(u >> 16);
}

__device__ __forceinline__ unsigned short sign_bf16(float w) {
    // sign() -> exact bf16 constants: +1.0 = 0x3F80, -1.0 = 0xBF80, 0 = 0x0000
    return w > 0.0f ? (unsigned short)0x3F80u
                    : (w < 0.0f ? (unsigned short)0xBF80u : (unsigned short)0u);
}

// One-time weight binarize + K-reorder: pw[oc*2304 + r*256 + ic] =
// sign(w[oc*2304 + ic*9 + r]).  589,824 elements, coalesced writes.
__global__ __launch_bounds__(256)
void pack_weights_kernel(const float* __restrict__ weight,
                         unsigned short* __restrict__ pw)
{
    const int i  = blockIdx.x * 256 + threadIdx.x;   // [0, 589824)
    const int oc = i / KK;
    const int rm = i - oc * KK;
    const int r  = rm >> 8;          // tap index kh*3+kw
    const int ic = rm & 255;
    pw[i] = sign_bf16(weight[oc * KK + ic * 9 + r]);
}

template <bool PACKED>
__global__ __launch_bounds__(256)
void binconv_wmma_kernel(const float* __restrict__ x,
                         const float* __restrict__ weight,
                         const unsigned short* __restrict__ pw,
                         const float* __restrict__ bias,
                         float* __restrict__ out)
{
    // ping-pong LDS double buffers (one barrier per slice)
    __shared__ __align__(16) unsigned short sAw[2][TILE_OC * LDS_STRIDE];
    __shared__ __align__(16) unsigned short sBx[2][TILE_PX * LDS_STRIDE];

    const int bx      = blockIdx.x;
    const int pixTile = bx % (NPIX / TILE_PX);   // 784 tiles
    const int ocTile  = bx / (NPIX / TILE_PX);   // 2 tiles
    const int pixOff  = pixTile * TILE_PX;
    const int ocOff   = ocTile * TILE_OC;

    const int tid   = threadIdx.x;
    const int wave  = tid >> 5;
    const int lane  = tid & 31;
    const int ln    = lane & 15;
    const int half  = lane >> 4;       // 0: lanes 0-15, 1: lanes 16-31
    const int waveM = wave & 3;        // 4 waves along oc  (32 oc each)
    const int waveN = wave >> 2;       // 2 waves along pix (64 pix each)

    // ---- per-thread staging roles (constant across whole kernel) ----
    // Bx gather: thread t -> pixel p = t&127, K half-slice = t>>7 (16 ic values)
    const int p     = tid & 127;
    const int khalf = tid >> 7;
    const int pix   = pixOff + p;
    const int nb_g  = pix / PIX;
    const int pp_g  = pix - nb_g * PIX;
    const int oh    = pp_g / W;
    const int ow    = pp_g - oh * W;
    const long xbase = (long)nb_g * (IN_CH * PIX);

    // Aw gather: thread t -> oc row = t>>1, 16 ic values at (t&1)*16
    const int wrow = tid >> 1;
    const int wc   = tid & 1;
    const unsigned short* pwRow =
        PACKED ? pw + (long)(ocOff + wrow) * KK + wc * 16 : nullptr;

    // Gather one 32-wide K slice (global -> regs).  s -> (tap r, ic base).
    auto gatherSlice = [&](int s, v8u16& wlo, v8u16& whi,
                           v8u16& xlo, v8u16& xhi) {
        const int r   = s >> 3;
        const int icb = (s & 7) << 5;

        if constexpr (PACKED) {
            const v8u16* src = (const v8u16*)(pwRow + r * 256 + icb);
            wlo = src[0];
            whi = src[1];
        } else {
            const float* wp = weight + (long)(ocOff + wrow) * KK + r;
            __align__(16) unsigned short ws16[16];
#pragma unroll
            for (int kk = 0; kk < 16; ++kk)
                ws16[kk] = sign_bf16(wp[(icb + wc * 16 + kk) * 9]);
            wlo = *(const v8u16*)&ws16[0];
            whi = *(const v8u16*)&ws16[8];
        }

        const int kh = r / 3;
        const int kw = r - kh * 3;
        const int ih = oh + kh - 1;
        const int iw = ow + kw - 1;
        const bool valid = ((unsigned)ih < (unsigned)H) &&
                           ((unsigned)iw < (unsigned)W);
        const float* xp = x + xbase + (long)ih * W + iw;   // deref only if valid
        __align__(16) unsigned short xb[16];
#pragma unroll
        for (int kk = 0; kk < 16; ++kk) {
            const int ic = icb + khalf * 16 + kk;
            const float v = valid ? xp[(long)ic * PIX] : 0.0f;
            xb[kk] = f32_to_bf16(v);
        }
        xlo = *(const v8u16*)&xb[0];
        xhi = *(const v8u16*)&xb[8];
    };

    v8f acc[2][4];
#pragma unroll
    for (int mi = 0; mi < 2; ++mi)
#pragma unroll
        for (int j = 0; j < 4; ++j) acc[mi][j] = (v8f)0.0f;

    v8u16 cw0, cw1, cx0, cx1;
    gatherSlice(0, cw0, cw1, cx0, cx1);                 // pipeline prologue

    for (int s = 0; s < NSLICE; ++s) {
        unsigned short* aw = &sAw[s & 1][0];
        unsigned short* bxs = &sBx[s & 1][0];

        // ---------- LDS stage (current slice) ----------
        v8u16* adst = (v8u16*)&aw[wrow * LDS_STRIDE + wc * 16];
        adst[0] = cw0;
        adst[1] = cw1;
        v8u16* bdst = (v8u16*)&bxs[p * LDS_STRIDE + khalf * 16];
        bdst[0] = cx0;
        bdst[1] = cx1;

        __syncthreads();   // buffer (s&1) ready; prior reads of it retired

        // ---------- issue next slice's global loads (overlap with WMMA) ----
        if (s + 1 < NSLICE)
            gatherSlice(s + 1, cw0, cw1, cx0, cx1);

        // ---------- WMMA: 2 (oc) x 4 (pix) fragments per wave ----------
        // A fragment (16x32 bf16): lane m = ln; half=0 holds K 0..15,
        // half=1 holds K 16..31 (same order as B).
        FragAB afr[2];
#pragma unroll
        for (int mi = 0; mi < 2; ++mi) {
            const unsigned short* aP =
                &aw[(waveM * 32 + mi * 16 + ln) * LDS_STRIDE + half * 16];
            afr[mi].h[0] = *(const v8u16*)aP;          // K = 16*half .. +7
            afr[mi].h[1] = *(const v8u16*)(aP + 8);    // K = 16*half + 8 .. +15
        }
#pragma unroll
        for (int j = 0; j < 4; ++j) {
            // B fragment (32x16 bf16): lane n = ln (pixel col);
            // half=0 -> K 0..15, half=1 -> K 16..31; contiguous 32B per lane.
            FragAB bfr;
            bfr.u = *(const v16u16*)
                &bxs[(waveN * 64 + j * 16 + ln) * LDS_STRIDE + half * 16];
#pragma unroll
            for (int mi = 0; mi < 2; ++mi) {
                acc[mi][j] = __builtin_amdgcn_wmma_f32_16x16x32_bf16(
                    false, afr[mi].b, false, bfr.b, (short)0,
                    acc[mi][j], false, false);
            }
        }
    }

    // ---------- epilogue: bias + coalesced stores ----------
    // C/D layout: lane n = ln (pixel), VGPR v -> oc row v (+8 for upper half).
#pragma unroll
    for (int mi = 0; mi < 2; ++mi) {
#pragma unroll
        for (int j = 0; j < 4; ++j) {
            const int pixc = pixOff + waveN * 64 + j * 16 + ln;
            const int nb   = pixc / PIX;
            const int pp   = pixc - nb * PIX;
            const long ob  = (long)nb * (OUT_CH * PIX) + pp;
#pragma unroll
            for (int v = 0; v < 8; ++v) {
                const int oc = ocOff + waveM * 32 + mi * 16 + (half ? v + 8 : v);
                out[ob + (long)oc * PIX] = acc[mi][j][v] + bias[oc];
            }
        }
    }
}

extern "C" void kernel_launch(void* const* d_in, const int* in_sizes, int n_in,
                              void* d_out, int out_size, void* d_ws, size_t ws_size,
                              hipStream_t stream) {
    (void)in_sizes; (void)n_in; (void)out_size;
    const float* x      = (const float*)d_in[0];
    const float* weight = (const float*)d_in[1];
    const float* bias   = (const float*)d_in[2];
    float* out          = (float*)d_out;

    const int grid = (NPIX / TILE_PX) * (OUT_CH / TILE_OC);  // 784 * 2 = 1568
    const size_t PW_BYTES = (size_t)OUT_CH * KK * sizeof(unsigned short); // 1.13 MB

    if (d_ws != nullptr && ws_size >= PW_BYTES) {
        unsigned short* pw = (unsigned short*)d_ws;
        pack_weights_kernel<<<(OUT_CH * KK) / 256, 256, 0, stream>>>(weight, pw);
        binconv_wmma_kernel<true><<<grid, 256, 0, stream>>>(x, weight, pw, bias, out);
    } else {
        binconv_wmma_kernel<false><<<grid, 256, 0, stream>>>(x, weight, nullptr, bias, out);
    }
}
